// FBPINN_PoU_7164005450317
// MI455X (gfx1250) — matrix operand order repaired and checked by reading, not compiled, not run on hardware
//
#include <hip/hip_runtime.h>
#include <hip/hip_bf16.h>

// ---------------------------------------------------------------------------
// FBPINN forward, fused: 64 subdomain MLPs (1->64->64->64->1, tanh) + Gaussian
// partition-of-unity blend + hard-BC ansatz, on the gfx1250 WMMA f16 path.
//
// S=64 subdomains, N=65536 points, W=64 width.
// Dominant work: 2 x (per-s [N,64]x[64,64] GEMM) ~= 69 GFLOP, compute-bound.
// Each block: 256 points, loops over all 64 subdomains; h stays in LDS.
// Each wave owns 32 points (two 16-col tiles) so every A (weight) fragment
// register is reused by two WMMAs -> halves LDS reads on the critical path.
// ---------------------------------------------------------------------------

typedef __attribute__((ext_vector_type(16))) _Float16 v16h;
typedef __attribute__((ext_vector_type(8)))  float    v8f;
typedef __attribute__((ext_vector_type(4)))  unsigned int u32x4;
typedef __attribute__((ext_vector_type(2)))  unsigned int u32x2;

#define NSUB   64
#define WID    64
#define PTS    256        // points per block (8 waves x 32 points)
#define PITCH  72         // halves per row in LDS (144B rows, bank-skewed)

__device__ __forceinline__ float tanh_fast(float v) {
    // tanh(x) = 1 - 2/(exp(2x)+1): v_exp_f32 + v_add + v_rcp_f32 + v_fma.
    float e = __expf(2.0f * v);
    return 1.0f - 2.0f * __builtin_amdgcn_rcpf(e + 1.0f);
}

union Frag32 { u32x4 q[2]; v16h v; };
union Pack16 { u32x4 q;    _Float16 h[8]; };
union Pack4  { u32x2 q;    _Float16 h[4]; };

#define WMMA_F16(A, B, C) \
    __builtin_amdgcn_wmma_f32_16x16x32_f16(false, (A), false, (B), (short)0, (C), false, false)

// One dense layer: hout[p][v] = tanh( sum_w Wl[v][w] * hin[p][w] + bias[v] )
// D = A x B; A = weight tile [16 out x 32 K]; B = h tile [32 K x 16 pts].
// Two point-tiles per wave: each A fragment feeds two WMMAs.
__device__ __forceinline__ void layer_wmma(const _Float16* __restrict__ Wl,
                                           const float*    __restrict__ bias,
                                           const _Float16* __restrict__ hin,
                                           _Float16*       __restrict__ hout,
                                           int wave, int lane)
{
    const int col = lane & 15;     // B column / A row-within-tile / D column
    const int hi  = lane >> 4;     // upper half-wave flag
    const int pA  = wave * 32 + col;   // point, tile 0
    // tile 1 point = pA + 16

    // B fragments: [point-tile][k-step]. elem e -> K = e + 16*hi + 32*k.
    Frag32 bfrag[2][2];
#pragma unroll
    for (int t = 0; t < 2; ++t) {
#pragma unroll
        for (int k = 0; k < 2; ++k) {
            const int K0 = 32 * k + 16 * hi;
            const u32x4* src = (const u32x4*)(hin + (pA + 16 * t) * PITCH + K0);
            bfrag[t][k].q[0] = src[0];
            bfrag[t][k].q[1] = src[1];
        }
    }

#pragma unroll
    for (int o = 0; o < 4; ++o) {          // 4 output tiles of 16 features
        const int row     = o * 16 + col;  // A-matrix row = output feature
        const int outbase = o * 16 + 8 * hi;

        // C init = bias broadcast. D layout: elem r -> out = 16o + r + 8*hi.
        v8f c0, c1;
#pragma unroll
        for (int r = 0; r < 8; ++r) {
            const float b = bias[outbase + r];
            c0[r] = b; c1[r] = b;
        }

#pragma unroll
        for (int k = 0; k < 2; ++k) {
            // A layout: elem e -> K = e + 16*hi + 32*k
            const int K0 = 32 * k + 16 * hi;
            Frag32 a;
            a.q[0] = *(const u32x4*)(Wl + row * PITCH + K0);
            a.q[1] = *(const u32x4*)(Wl + row * PITCH + K0 + 8);
            c0 = WMMA_F16(a.v, bfrag[0][k].v, c0);   // A register reused
            c1 = WMMA_F16(a.v, bfrag[1][k].v, c1);
        }

        // tanh, pack to f16, one 16B store per tile
        Pack16 d0, d1;
#pragma unroll
        for (int r = 0; r < 8; ++r) {
            d0.h[r] = (_Float16)tanh_fast(c0[r]);
            d1.h[r] = (_Float16)tanh_fast(c1[r]);
        }
        *(u32x4*)(hout + pA * PITCH + outbase)        = d0.q;
        *(u32x4*)(hout + (pA + 16) * PITCH + outbase) = d1.q;
    }
}

__global__ __launch_bounds__(256) void fbpinn_fused_kernel(
    const float* __restrict__ x,
    const float* __restrict__ W0, const float* __restrict__ b0,
    const float* __restrict__ W1, const float* __restrict__ b1,
    const float* __restrict__ W2, const float* __restrict__ b2,
    const float* __restrict__ W3, const float* __restrict__ b3,
    float* __restrict__ out, int N)
{
    __shared__ __align__(16) _Float16 sW1[WID * PITCH];   //  9.0 KB
    __shared__ __align__(16) _Float16 sW2[WID * PITCH];   //  9.0 KB
    __shared__ __align__(16) _Float16 sHA[PTS * PITCH];   // 36.0 KB
    __shared__ __align__(16) _Float16 sHB[PTS * PITCH];   // 36.0 KB
    __shared__ __align__(16) _Float16 sW3h[WID];          // f16 W3 row
    __shared__ float sW0v[WID], sB0[WID], sB1[WID], sB2[WID];
    __shared__ float sB3[NSUB];
    __shared__ float sX[PTS], sAcc[PTS], sDen[PTS];

    const int tid  = threadIdx.x;
    const int wave = tid >> 5;
    const int lane = tid & 31;
    const int col  = lane & 15;
    const int hi   = lane >> 4;
    const int p0   = blockIdx.x * PTS;

    const float inv_sigma = (float)NSUB / 1.5f;   // 1/sigma, sigma = 1.5/S

    // --- per-point init: x, accumulator, PoU denominator (s-independent) ---
    {
        const int g = p0 + tid;
        float xv = (g < N) ? x[g] : 0.0f;
        sX[tid]   = xv;
        sAcc[tid] = 0.0f;
        float den = 0.0f;
#pragma unroll 4
        for (int s = 0; s < NSUB; ++s) {
            float d = (xv - (float)s * (1.0f / 63.0f)) * inv_sigma;
            den += __expf(-0.5f * d * d);
        }
        sDen[tid] = den;
    }
    if (tid < NSUB) sB3[tid] = b3[tid];
    __syncthreads();

#pragma unroll 1
    for (int s = 0; s < NSUB; ++s) {
        // ---- stage weights into LDS (f32 -> f16), 128-bit global loads ----
        const float4* g1 = (const float4*)(W1 + s * WID * WID);
        const float4* g2 = (const float4*)(W2 + s * WID * WID);
        for (int i = tid; i < (WID * WID) / 4; i += 256) {   // 1024 float4s
            const int r = i >> 4;            // 16 float4 per 64-wide row
            const int c = (i & 15) * 4;
            float4 w1 = g1[i];
            float4 w2 = g2[i];
            Pack4 pk;
            pk.h[0] = (_Float16)w1.x; pk.h[1] = (_Float16)w1.y;
            pk.h[2] = (_Float16)w1.z; pk.h[3] = (_Float16)w1.w;
            *(u32x2*)(sW1 + r * PITCH + c) = pk.q;
            pk.h[0] = (_Float16)w2.x; pk.h[1] = (_Float16)w2.y;
            pk.h[2] = (_Float16)w2.z; pk.h[3] = (_Float16)w2.w;
            *(u32x2*)(sW2 + r * PITCH + c) = pk.q;
        }
        if (tid < WID) {
            sW0v[tid] = W0[s * WID + tid];
            sB0[tid]  = b0[s * WID + tid];
            sB1[tid]  = b1[s * WID + tid];
            sB2[tid]  = b2[s * WID + tid];
            sW3h[tid] = (_Float16)W3[s * WID + tid];
        }
        // prefetch next subdomain's weights into cache (global_prefetch_b8)
        if (s + 1 < NSUB) {
            __builtin_prefetch((const float*)g1 + WID * WID + tid * 16, 0, 1);
            __builtin_prefetch((const float*)g2 + WID * WID + tid * 16, 0, 1);
        }
        __syncthreads();

        // ---- layer 0 (rank-1): h0[p][w] = tanh(W0[w]*x[p] + b0[w]) ----
        {
            const float xv = sX[tid];
            _Float16* hp = sHA + tid * PITCH;
#pragma unroll
            for (int w = 0; w < WID; w += 2) {
                float h0 = tanh_fast(fmaf(sW0v[w],     xv, sB0[w]));
                float h1 = tanh_fast(fmaf(sW0v[w + 1], xv, sB0[w + 1]));
                union { _Float16 h[2]; unsigned u; } pk;
                pk.h[0] = (_Float16)h0;
                pk.h[1] = (_Float16)h1;
                *(unsigned*)(hp + w) = pk.u;
            }
        }
        __syncthreads();

        // ---- layers 1 & 2: WMMA GEMM + tanh, ping-pong through LDS ----
        layer_wmma(sW1, sB1, sHA, sHB, wave, lane);
        __syncthreads();
        layer_wmma(sW2, sB2, sHB, sHA, wave, lane);
        // No barrier: layer 2 stored only this wave's 32 points, and
        // same-wave DS ops are ordered; layer 3 reads those columns back.

        // ---- layer 3 as WMMA: u[p] = W3 . h2[p] + b3  (row 0 of D) ----
        {
            const int pA = wave * 32 + col;
            v8f c3a = {}, c3b = {};
#pragma unroll
            for (int k = 0; k < 2; ++k) {
                const int K0b = 32 * k + 16 * hi;
                Frag32 bfA, bfB;
                const u32x4* srcA = (const u32x4*)(sHA + pA * PITCH + K0b);
                const u32x4* srcB = (const u32x4*)(sHA + (pA + 16) * PITCH + K0b);
                bfA.q[0] = srcA[0]; bfA.q[1] = srcA[1];
                bfB.q[0] = srcB[0]; bfB.q[1] = srcB[1];

                Frag32 a3 = {};                // rows 1..15 stay zero
                if (col == 0) {                // lane 0 / lane 16 carry row 0
                    const int K0a = 32 * k + 16 * hi;
                    a3.q[0] = *(const u32x4*)(sW3h + K0a);
                    a3.q[1] = *(const u32x4*)(sW3h + K0a + 8);
                }
                c3a = WMMA_F16(a3.v, bfA.v, c3a);
                c3b = WMMA_F16(a3.v, bfB.v, c3b);
            }
            // D elem 0, lanes 0..15: M=0 row -> u for this wave's points
            if (lane < 16) {
                const float b3v = sB3[s];
                const float cs  = (float)s * (1.0f / 63.0f);
                const int pa = wave * 32 + lane;
                const int pb = pa + 16;
                float da = (sX[pa] - cs) * inv_sigma;
                float db = (sX[pb] - cs) * inv_sigma;
                sAcc[pa] = fmaf(__expf(-0.5f * da * da), c3a[0] + b3v, sAcc[pa]);
                sAcc[pb] = fmaf(__expf(-0.5f * db * db), c3b[0] + b3v, sAcc[pb]);
            }
        }
        __syncthreads();   // protect weight/h buffers before next s
    }

    // ---- epilogue: hard-BC ansatz tanh(5x) * (sum raw_s u_s)/den ----
    {
        const int g = p0 + tid;
        if (g < N) {
            float xv = sX[tid];
            out[g] = tanh_fast(5.0f * xv) *
                     (sAcc[tid] * __builtin_amdgcn_rcpf(sDen[tid]));
        }
    }
}

extern "C" void kernel_launch(void* const* d_in, const int* in_sizes, int n_in,
                              void* d_out, int out_size, void* d_ws, size_t ws_size,
                              hipStream_t stream) {
    const float* x  = (const float*)d_in[0];
    const float* W0 = (const float*)d_in[1];
    const float* b0 = (const float*)d_in[2];
    const float* W1 = (const float*)d_in[3];
    const float* b1 = (const float*)d_in[4];
    const float* W2 = (const float*)d_in[5];
    const float* b2 = (const float*)d_in[6];
    const float* W3 = (const float*)d_in[7];
    const float* b3 = (const float*)d_in[8];
    float* out = (float*)d_out;

    const int N = in_sizes[0];                 // 65536 collocation points
    const int nblocks = (N + PTS - 1) / PTS;   // 256 blocks of 256 threads

    hipLaunchKernelGGL(fbpinn_fused_kernel, dim3(nblocks), dim3(256), 0, stream,
                       x, W0, b0, W1, b1, W2, b2, W3, b3, out, N);
}
